// REMSA_15401752723494
// MI455X (gfx1250) — matrix operand order repaired and checked by reading, not compiled, not run on hardware
//
#include <hip/hip_runtime.h>

typedef __bf16 bf16;
typedef __attribute__((ext_vector_type(16))) __bf16 v16bf;
typedef __attribute__((ext_vector_type(8)))  float  v8f;
typedef __attribute__((ext_vector_type(4)))  unsigned int v4u;
typedef __attribute__((ext_vector_type(8)))  int v8i;
typedef __attribute__((ext_vector_type(4)))  int v4i;

#define MODE_F32  0
#define MODE_BF16 1
#define MODE_ACC  2

#if defined(__has_builtin)
#if __has_builtin(__builtin_amdgcn_tensor_load_to_lds) && \
    __has_builtin(__builtin_amdgcn_s_wait_tensorcnt)
#define HAVE_TDM 1
#endif
#endif

// ---------------------------------------------------------------- conversions
__global__ void cvt_f32_to_bf16_v8(const float* __restrict__ in, bf16* __restrict__ out, int n) {
    int i = (blockIdx.x * 256 + threadIdx.x) * 8;
    if (i < n) {
        float4 f0 = *(const float4*)(in + i);
        float4 f1 = *(const float4*)(in + i + 4);
        union { uint4 u; bf16 e[8]; } o;
        o.e[0] = (bf16)f0.x; o.e[1] = (bf16)f0.y; o.e[2] = (bf16)f0.z; o.e[3] = (bf16)f0.w;
        o.e[4] = (bf16)f1.x; o.e[5] = (bf16)f1.y; o.e[6] = (bf16)f1.z; o.e[7] = (bf16)f1.w;
        *(uint4*)(out + i) = o.u;
    }
}

// in[r][c] (rows x cols, f32) -> out[c][r] (bf16); division-free 2D indexing
__global__ void cvt_transpose_bf16(const float* __restrict__ in, bf16* __restrict__ out,
                                   int rows, int cols) {
    int c = blockIdx.x * 256 + threadIdx.x;
    int r = blockIdx.y;
    if (c < cols) out[(size_t)c * rows + r] = (bf16)in[(size_t)r * cols + c];
}

// ---------------------------------------------------------------- TDM helper
#ifdef HAVE_TDM
// Load a 128-row x 32-element bf16 tile (global row stride = strideElems) into LDS at
// lds_off, padding 4 dwords after every 16 data dwords => 80B LDS row stride ([40] arrays).
__device__ __forceinline__ void tdm_tile_load(unsigned lds_off, const bf16* gptr,
                                              unsigned strideElems, unsigned td0, unsigned td1) {
    unsigned long long ga = (unsigned long long)(size_t)gptr;
    v4u g0 = {1u,                                        // count=1, user descriptor
              lds_off,                                   // lds_addr (bytes)
              (unsigned)ga,                              // global_addr[31:0]
              ((unsigned)(ga >> 32) & 0x01FFFFFFu) | 0x80000000u}; // addr[56:32] | type=2
    v8i g1;
    g1[0] = (int)((1u << 16)        // data_size = 1 (2 bytes)
                | (1u << 20)        // pad_enable
                | (3u << 22)        // pad_interval: 16 data dwords (64B)
                | (3u << 25));      // pad_amount: 4 dwords (16B)
    g1[1] = (int)((td0 & 0xFFFFu) << 16);                // tensor_dim0[15:0]
    g1[2] = (int)((td0 >> 16) | ((td1 & 0xFFFFu) << 16));// tensor_dim0[31:16], dim1[15:0]
    g1[3] = (int)((td1 >> 16) | (32u << 16));            // dim1[31:16], tile_dim0=32
    g1[4] = (int)128u;                                   // tile_dim1=128, tile_dim2=0
    g1[5] = (int)strideElems;                            // tensor_dim0_stride[31:0]
    g1[6] = 0;
    g1[7] = 0;
    v4i z4 = {0, 0, 0, 0};
#if __clang_major__ >= 23
    v8i z8 = {0, 0, 0, 0, 0, 0, 0, 0};
    __builtin_amdgcn_tensor_load_to_lds(g0, g1, z4, z4, z8, 0);
#else
    __builtin_amdgcn_tensor_load_to_lds(g0, g1, z4, z4, 0);
#endif
}
#endif

// ---------------------------------------------------------------- WMMA GEMM
// C[M,N] = A[M,K] @ Bt[N,K]^T + bias[N] ; A,Bt bf16 ; f32 accum; MODE compile-time.
template <int MODE>
__global__ __launch_bounds__(256, 1)
void gemm_bf16_wmma(const bf16* __restrict__ A, const bf16* __restrict__ Bt,
                    const float* __restrict__ bias, float* __restrict__ outF,
                    bf16* __restrict__ outH, int M, int N, int K) {
    __shared__ __align__(16) bf16 As[2][128][40];  // [buf][row][k]  80B row stride
    __shared__ __align__(16) bf16 Bs[2][128][40];  // [buf][col][k]

    const int tid  = threadIdx.x;
    const int lane = tid & 31;
    const int w    = tid >> 5;
    const int hf   = lane >> 4;
    const int lm   = lane & 15;

    const int blockM = blockIdx.y * 128;
    const int blockN = blockIdx.x * 128;
    const int waveM  = (w >> 1) * 32;
    const int waveN  = (w & 1) * 64;

    const v8f zero8 = {0.f, 0.f, 0.f, 0.f, 0.f, 0.f, 0.f, 0.f};
    v8f acc[2][4];
#pragma unroll
    for (int mi = 0; mi < 2; ++mi)
#pragma unroll
        for (int ni = 0; ni < 4; ++ni) acc[mi][ni] = zero8;

    const int nk = K >> 5;

#ifdef HAVE_TDM
    const bf16* aBase = A  + (size_t)blockM * K;
    const bf16* bBase = Bt + (size_t)blockN * K;
    if (tid == 0) {
        tdm_tile_load((unsigned)(size_t)&As[0][0][0], aBase, (unsigned)K, (unsigned)K, (unsigned)M);
        tdm_tile_load((unsigned)(size_t)&Bs[0][0][0], bBase, (unsigned)K, (unsigned)K, (unsigned)N);
    }
    for (int i = 0; i < nk; ++i) {
        const int cur = i & 1;
        if (tid == 0) {
            if (i + 1 < nk) {
                tdm_tile_load((unsigned)(size_t)&As[cur ^ 1][0][0], aBase + (i + 1) * 32,
                              (unsigned)K, (unsigned)K, (unsigned)M);
                tdm_tile_load((unsigned)(size_t)&Bs[cur ^ 1][0][0], bBase + (i + 1) * 32,
                              (unsigned)K, (unsigned)K, (unsigned)N);
                __builtin_amdgcn_s_wait_tensorcnt(2);  // current tile's 2 loads complete
            } else {
                __builtin_amdgcn_s_wait_tensorcnt(0);
            }
        }
        __syncthreads();  // tile visible to all waves
#else
    const int aRow = tid >> 1, aKh = (tid & 1) * 16;  // mirrored for A and Bt
    for (int i = 0; i < nk; ++i) {
        const int cur = 0;
        const int k0 = i * 32;
        const bf16* ag = A  + (size_t)(blockM + aRow) * K + k0 + aKh;
        uint4 a0 = *(const uint4*)ag;
        uint4 a1 = *(const uint4*)(ag + 8);
        const bf16* bg = Bt + (size_t)(blockN + aRow) * K + k0 + aKh;
        uint4 b0 = *(const uint4*)bg;
        uint4 b1 = *(const uint4*)(bg + 8);
        __syncthreads();
        *(uint4*)&As[0][aRow][aKh]     = a0;
        *(uint4*)&As[0][aRow][aKh + 8] = a1;
        *(uint4*)&Bs[0][aRow][aKh]     = b0;
        *(uint4*)&Bs[0][aRow][aKh + 8] = b1;
        __syncthreads();
#endif
        // fragments (ISA 16-bit layouts)
        v16bf af[2], bfv[4];
#pragma unroll
        for (int mi = 0; mi < 2; ++mi) {
            const bf16* ap = &As[cur][waveM + mi * 16 + lm][hf * 8];
            ((uint4*)&af[mi])[0] = *(const uint4*)ap;
            ((uint4*)&af[mi])[1] = *(const uint4*)(ap + 16);
        }
#pragma unroll
        for (int ni = 0; ni < 4; ++ni) {
            const bf16* bp = &Bs[cur][waveN + ni * 16 + lm][hf * 8];
            ((uint4*)&bfv[ni])[0] = *(const uint4*)bp;
            ((uint4*)&bfv[ni])[1] = *(const uint4*)(bp + 16);
        }
#pragma unroll
        for (int mi = 0; mi < 2; ++mi)
#pragma unroll
            for (int ni = 0; ni < 4; ++ni)
                acc[mi][ni] = __builtin_amdgcn_wmma_f32_16x16x32_bf16(
                    false, af[mi], false, bfv[ni], (short)0, acc[mi][ni], false, false);
        __syncthreads();  // all waves done reading before buffer reuse
    }

    // epilogue: C/D layout => M = r + 8*hf, N = lane&15
#pragma unroll
    for (int ni = 0; ni < 4; ++ni) {
        const int col = blockN + waveN + ni * 16 + lm;
        const float bv = bias[col];
#pragma unroll
        for (int mi = 0; mi < 2; ++mi) {
            const int row0 = blockM + waveM + mi * 16 + hf * 8;
#pragma unroll
            for (int r = 0; r < 8; ++r) {
                const size_t idx = (size_t)(row0 + r) * N + col;
                const float v = acc[mi][ni][r] + bv;
                if (MODE == MODE_F32)       outF[idx] = v;
                else if (MODE == MODE_BF16) outH[idx] = (bf16)v;
                else                        outF[idx] += v;
            }
        }
    }
}

// ---------------------------------------------------------------- attention
// qkv row per token: [h][q(64) k(64) v(64)], stride 192/head. One wave per token.
__global__ __launch_bounds__(256, 1)
void attn_wmma(const bf16* __restrict__ qkv, const float* __restrict__ pb,
               bf16* __restrict__ ctx, int Nseq /* power of two */) {
    __shared__ __align__(16) bf16 vt[8][64][24];  // per-wave v^T [d][g]
    __shared__ __align__(16) bf16 wl[8][16][16];  // per-wave softmax weights [h][g]

    const int tid  = threadIdx.x;
    const int lane = tid & 31;
    const int w    = tid >> 5;
    const int hf   = lane >> 4;
    const int lm   = lane & 15;

    const unsigned tt = blockIdx.x * 8u + (unsigned)w;
    const unsigned n  = tt & (unsigned)(Nseq - 1);   // Nseq is a power of two
    const bf16* row = qkv + (size_t)tt * 3072;

    {   // stage v transposed: g = lm, d-half = 32*hf
        const bf16* vp = row + lm * 192 + 128 + hf * 32;
        union { uint4 u4[4]; bf16 e[32]; } tv;
        tv.u4[0] = *(const uint4*)(vp);
        tv.u4[1] = *(const uint4*)(vp + 8);
        tv.u4[2] = *(const uint4*)(vp + 16);
        tv.u4[3] = *(const uint4*)(vp + 24);
#pragma unroll
        for (int e = 0; e < 32; ++e) vt[w][hf * 32 + e][lm] = tv.e[e];
    }

    // scores = q @ k^T  (K=64 in two 32-chunks)
    v8f sc = {0.f, 0.f, 0.f, 0.f, 0.f, 0.f, 0.f, 0.f};
#pragma unroll
    for (int c = 0; c < 64; c += 32) {
        v16bf qa, kb;
        const bf16* qp = row + lm * 192 + c + hf * 8;
        ((uint4*)&qa)[0] = *(const uint4*)qp;
        ((uint4*)&qa)[1] = *(const uint4*)(qp + 16);
        const bf16* kp = row + lm * 192 + 64 + c + hf * 8;
        ((uint4*)&kb)[0] = *(const uint4*)kp;
        ((uint4*)&kb)[1] = *(const uint4*)(kp + 16);
        sc = __builtin_amdgcn_wmma_f32_16x16x32_bf16(false, qa, false, kb, (short)0,
                                                     sc, false, false);
    }

    // softmax over g (lane axis within 16-lane half); h = r + 8*hf
    const float* pbrow = pb + (size_t)n * 256;
#pragma unroll
    for (int r = 0; r < 8; ++r) {
        const int h = r + hf * 8;
        float v = sc[r] * 0.125f + pbrow[h * 16 + lm];
        float m = v;
        m = fmaxf(m, __shfl_xor(m, 1, 32));
        m = fmaxf(m, __shfl_xor(m, 2, 32));
        m = fmaxf(m, __shfl_xor(m, 4, 32));
        m = fmaxf(m, __shfl_xor(m, 8, 32));
        float e = __expf(v - m);
        float s = e;
        s += __shfl_xor(s, 1, 32);
        s += __shfl_xor(s, 2, 32);
        s += __shfl_xor(s, 4, 32);
        s += __shfl_xor(s, 8, 32);
        wl[w][h][lm] = (bf16)(e / s);
    }

    asm volatile("s_wait_dscnt 0" ::: "memory");  // wave-private LDS RAW

    v16bf wa;  // A fragment (K padded 16->32: upper slots zero)
    {
        const uint4 z4 = {0u, 0u, 0u, 0u};
        ((uint4*)&wa)[0] = *(const uint4*)&wl[w][lm][hf * 8];
        ((uint4*)&wa)[1] = z4;
    }

#pragma unroll
    for (int c = 0; c < 4; ++c) {  // attn = W @ V, 4 d-chunks
        v16bf vb;
        {
            const bf16* vp = &vt[w][c * 16 + lm][hf * 8];
            const uint4 z4 = {0u, 0u, 0u, 0u};
            ((uint4*)&vb)[0] = *(const uint4*)vp;
            ((uint4*)&vb)[1] = z4;
        }
        v8f o = {0.f, 0.f, 0.f, 0.f, 0.f, 0.f, 0.f, 0.f};
        o = __builtin_amdgcn_wmma_f32_16x16x32_bf16(false, wa, false, vb, (short)0,
                                                    o, false, false);
#pragma unroll
        for (int r = 0; r < 8; ++r)
            ctx[(size_t)tt * 1024 + (r + hf * 8) * 64 + c * 16 + lm] = (bf16)o[r];
    }
}

// ---------------------------------------------------------------- depthwise conv (k=3, pad=1)
// C = 1024, Nseq = 4096 hardcoded -> pure shift/mask indexing, no division.
__global__ void dwconv3(const bf16* __restrict__ in, const float* __restrict__ wgt,
                        const float* __restrict__ bias, bf16* __restrict__ out, int total) {
    int i = blockIdx.x * 256 + threadIdx.x;
    if (i >= total) return;
    int c = i & 1023;           // channel  (C = 1024)
    int n = (i >> 10) & 4095;   // position (Nseq = 4096)
    float acc = bias[c];
    if (n > 0)    acc += (float)in[i - 1024] * wgt[c * 3 + 0];
    acc += (float)in[i] * wgt[c * 3 + 1];
    if (n < 4095) acc += (float)in[i + 1024] * wgt[c * 3 + 2];
    out[i] = (bf16)acc;
}

// ---------------------------------------------------------------- launch
extern "C" void kernel_launch(void* const* d_in, const int* in_sizes, int n_in,
                              void* d_out, int out_size, void* d_ws, size_t ws_size,
                              hipStream_t stream) {
    const int Bb = 4, Nseq = 4096, C = 1024;
    const int M = Bb * Nseq;  // 16384 tokens

    const float* x      = (const float*)d_in[0];
    const float* qkv_w  = (const float*)d_in[1];
    const float* qkv_b  = (const float*)d_in[2];
    const float* out_w  = (const float*)d_in[3];
    const float* out_b  = (const float*)d_in[4];
    const float* pos_b  = (const float*)d_in[5];
    const float* conv_w = (const float*)d_in[6];
    const float* conv_b = (const float*)d_in[7];
    const float* pw_w   = (const float*)d_in[8];
    const float* pw_b   = (const float*)d_in[9];
    const float* dw_w   = (const float*)d_in[10];
    const float* dw_b   = (const float*)d_in[11];
    const float* tok_w  = (const float*)d_in[12];
    const float* tok_b  = (const float*)d_in[13];
    float* outp = (float*)d_out;

    char* ws = (char*)d_ws;
    size_t off = 0;
    auto carve = [&](size_t bytes) -> char* {
        char* p = ws + off;
        off += (bytes + 255) & ~(size_t)255;
        return p;
    };
    bf16* x_bf    = (bf16*)carve((size_t)M * C * 2);
    bf16* qkv_wt  = (bf16*)carve((size_t)3 * C * C * 2);  // qkv_w^T  [3C][C]
    bf16* out_wt  = (bf16*)carve((size_t)C * C * 2);      // out_w^T
    bf16* conv_wt = (bf16*)carve((size_t)C * C * 2);      // conv_w^T
    bf16* pw_wb   = (bf16*)carve((size_t)C * C * 2);      // pw_w is already (out,in)
    bf16* tok_wt  = (bf16*)carve((size_t)C * C * 2);      // tok_w^T
    bf16* ctx_bf  = (bf16*)carve((size_t)M * C * 2);
    bf16* qkv_bf  = (bf16*)carve((size_t)M * 3 * C * 2);
    // after attention + out-gemm, qkv_bf is dead -> reuse for conv branch
    bf16* buf1 = qkv_bf;
    bf16* buf2 = qkv_bf + (size_t)M * C;
    bf16* buf3 = qkv_bf + (size_t)2 * M * C;

    dim3 blk(256);
    auto blocks  = [](int n) { return (n + 255) / 256; };
    auto blocks8 = [](int n) { return (n / 8 + 255) / 256; };

    // 1) conversions (weights pre-transposed to [N][K] for the GEMM B side)
    cvt_f32_to_bf16_v8<<<blocks8(M * C), blk, 0, stream>>>(x, x_bf, M * C);
    cvt_f32_to_bf16_v8<<<blocks8(C * C), blk, 0, stream>>>(pw_w, pw_wb, C * C);
    cvt_transpose_bf16<<<dim3((3 * C + 255) / 256, C), blk, 0, stream>>>(qkv_w, qkv_wt, C, 3 * C);
    cvt_transpose_bf16<<<dim3((C + 255) / 256, C), blk, 0, stream>>>(out_w, out_wt, C, C);
    cvt_transpose_bf16<<<dim3((C + 255) / 256, C), blk, 0, stream>>>(conv_w, conv_wt, C, C);
    cvt_transpose_bf16<<<dim3((C + 255) / 256, C), blk, 0, stream>>>(tok_w, tok_wt, C, C);

    // 2) qkv = x @ qkv_w + qkv_b  (bf16)
    gemm_bf16_wmma<MODE_BF16><<<dim3(3 * C / 128, M / 128), blk, 0, stream>>>(
        x_bf, qkv_wt, qkv_b, nullptr, qkv_bf, M, 3 * C, C);

    // 3) per-token attention -> ctx (bf16)
    attn_wmma<<<M / 8, blk, 0, stream>>>(qkv_bf, pos_b, ctx_bf, Nseq);

    // 4) attn_out = ctx @ out_w + out_b -> d_out (f32)
    gemm_bf16_wmma<MODE_F32><<<dim3(C / 128, M / 128), blk, 0, stream>>>(
        ctx_bf, out_wt, out_b, outp, nullptr, M, C, C);

    // 5) conv branch: xc = x @ conv_w + conv_b
    gemm_bf16_wmma<MODE_BF16><<<dim3(C / 128, M / 128), blk, 0, stream>>>(
        x_bf, conv_wt, conv_b, nullptr, buf1, M, C, C);

    // 6) pointwise conv (k=1): xc @ pw_w^T + pw_b
    gemm_bf16_wmma<MODE_BF16><<<dim3(C / 128, M / 128), blk, 0, stream>>>(
        buf1, pw_wb, pw_b, nullptr, buf2, M, C, C);

    // 7) depthwise conv k=3 pad=1
    dwconv3<<<blocks(M * C), blk, 0, stream>>>(buf2, dw_w, dw_b, buf3, M * C);

    // 8) tok gemm, accumulate into d_out
    gemm_bf16_wmma<MODE_ACC><<<dim3(C / 128, M / 128), blk, 0, stream>>>(
        buf3, tok_wt, tok_b, outp, nullptr, M, C, C);
}
